// SwinTransformerBlock_2164663517590
// MI455X (gfx1250) — matrix operand order repaired and checked by reading, not compiled, not run on hardware
//
#include <hip/hip_runtime.h>
#include <stdint.h>

// ---------------------------------------------------------------------------
// Swin block with Identity attention collapses to:
//   y1  = x + LN1(x)
//   out = y1 + GELU(LN2(y1) @ w1 + b1) @ w2 + b2
// B=32, L=4096 (H=W=64), C=256, HIDDEN=1024  -> 131072 rows.
// Strategy: bf16 WMMA (v_wmma_f32_16x16x32_bf16), fully fused per 128-row
// block; GELU intermediate (128x1024 bf16 = 256KB) lives entirely in LDS.
// ---------------------------------------------------------------------------

typedef __attribute__((ext_vector_type(16))) __bf16 v16bf;
typedef __attribute__((ext_vector_type(8)))  float  v8f;

#define C_DIM   256
#define HID     1024
#define MBLK    128
#define NROWS   131072          // 32 * 4096
#define LDS_BYTES 327680        // 64KB m-fragments + 256KB h-fragments

__device__ __forceinline__ unsigned short f2bf(float f) {
    union { float f; unsigned u; } c; c.f = f;
    unsigned u = c.u + 0x7FFFu + ((c.u >> 16) & 1u);   // round-to-nearest-even
    return (unsigned short)(u >> 16);
}

__device__ __forceinline__ float waveSum(float v) {
    #pragma unroll
    for (int m = 16; m >= 1; m >>= 1) v += __shfl_xor(v, m, 32);
    return v;
}

// Branch-free erf (Abramowitz & Stegun 7.1.26, |err| < 1.5e-7) -> exact-GELU.
// No divergent branches; v_rcp_f32 + v_exp_f32 are TRANS ops that co-execute
// with WMMA. (Correctly-rounded rcp would expand to a full v_div_scale
// sequence -- pointless given the approximation's own 1e-7 error.)
__device__ __forceinline__ float gelu_exact(float v) {
    float x = v * 0.70710678118f;
    float a = __builtin_fabsf(x);
    float t = __builtin_amdgcn_rcpf(__builtin_fmaf(0.3275911f, a, 1.0f));
    float p = __builtin_fmaf(1.061405429f, t, -1.453152027f);
    p = __builtin_fmaf(p, t,  1.421413741f);
    p = __builtin_fmaf(p, t, -0.284496736f);
    p = __builtin_fmaf(p, t,  0.254829592f);
    float e  = __expf(-a * a);
    float y  = __builtin_fmaf(-p * t, e, 1.0f);          // erf(|x|)
    float er = __builtin_copysignf(y, x);                // erf(x)
    return 0.5f * v * (1.0f + er);
}

// ---------------------------------------------------------------------------
// Pack f32 weights [K, N] row-major into bf16 WMMA B-fragments.
// Tile (kt, nt) covers K in [kt*32, kt*32+32), N in [nt*16, nt*16+16).
// B-matrix 32x16 bf16 layout (wave32): lane = (N%16) + 16*(((K-k0)>>3)&1),
// element e of a lane holds K = k0 + 8*(lane/16) + (e&7) + 2*(e&8). Storage: tile*512 + lane*16.
// ---------------------------------------------------------------------------
__global__ void pack_w_kernel(const float* __restrict__ w,
                              unsigned short* __restrict__ wp,
                              int K, int N) {
    int gid  = blockIdx.x * blockDim.x + threadIdx.x;
    int lane = gid & 31;
    int tile = gid >> 5;
    int NT   = N >> 4;
    int ntiles = (K >> 5) * NT;
    if (tile >= ntiles) return;
    int kt = tile / NT;
    int nt = tile - kt * NT;
    int n  = nt * 16 + (lane & 15);
    int kb = (lane & 16) ? 8 : 0;
    size_t base = (size_t)tile * 512 + (size_t)lane * 16;
    #pragma unroll
    for (int e = 0; e < 16; ++e) {
        int k = kt * 32 + kb + (e & 7) + ((e & 8) << 1);
        wp[base + e] = f2bf(w[(size_t)k * N + n]);
    }
}

// ---------------------------------------------------------------------------
// Fused kernel: one block = 128 rows, 256 threads (8 waves).
// ---------------------------------------------------------------------------
__global__ void __launch_bounds__(256)
swin_mlp_kernel(const float* __restrict__ x,
                const float* __restrict__ g1, const float* __restrict__ bt1,
                const float* __restrict__ g2, const float* __restrict__ bt2,
                const unsigned short* __restrict__ w1p,
                const float* __restrict__ b1,
                const unsigned short* __restrict__ w2p,
                const float* __restrict__ b2,
                float* __restrict__ out) {
    extern __shared__ char smem[];
    unsigned short* mBuf = (unsigned short*)smem;             // 64KB: LN2 out, A-frag layout
    unsigned short* hBuf = (unsigned short*)(smem + 65536);   // 256KB: GELU out, A-frag layout

    const int lane = threadIdx.x & 31;
    const int wv   = threadIdx.x >> 5;
    const int blk0 = blockIdx.x * MBLK;

    // ---------------- Phase A: LN1 + residual, LN2 -> LDS A-fragments -------
    {
        float gv1[8], bv1[8], gv2[8], bv2[8];
        #pragma unroll
        for (int j = 0; j < 8; ++j) {
            int c = lane * 8 + j;
            gv1[j] = g1[c]; bv1[j] = bt1[c];
            gv2[j] = g2[c]; bv2[j] = bt2[c];
        }
        // A-fragment scatter address components (fixed per lane):
        // this lane owns columns k = lane*8 .. lane*8+7 of each row.
        const int kt   = lane >> 2;                 // k / 32
        const int hiL  = lane & 1;                  // ((k%32)&8)>>3
        const int e0   = (lane & 2) ? 8 : 0;        // ((k%32)&16)>>1

        for (int i = 0; i < 16; ++i) {
            int mr  = wv * 16 + i;                  // row within block
            int row = blk0 + mr;
            const float4* xr = (const float4*)(x + (size_t)row * C_DIM + lane * 8);
            float4 a0 = xr[0], a1 = xr[1];
            float xv[8] = {a0.x, a0.y, a0.z, a0.w, a1.x, a1.y, a1.z, a1.w};

            float s = 0.f;
            #pragma unroll
            for (int j = 0; j < 8; ++j) s += xv[j];
            float mu = waveSum(s) * (1.0f / 256.0f);
            float vs = 0.f;
            #pragma unroll
            for (int j = 0; j < 8; ++j) { float d = xv[j] - mu; vs += d * d; }
            float rs = rsqrtf(waveSum(vs) * (1.0f / 256.0f) + 1e-5f);

            float y[8];
            #pragma unroll
            for (int j = 0; j < 8; ++j)
                y[j] = xv[j] + (xv[j] - mu) * rs * gv1[j] + bv1[j];

            float4 o0 = {y[0], y[1], y[2], y[3]};
            float4 o1 = {y[4], y[5], y[6], y[7]};
            float4* orow = (float4*)(out + (size_t)row * C_DIM + lane * 8);
            orow[0] = o0; orow[1] = o1;

            // LN2
            s = 0.f;
            #pragma unroll
            for (int j = 0; j < 8; ++j) s += y[j];
            float mu2 = waveSum(s) * (1.0f / 256.0f);
            vs = 0.f;
            #pragma unroll
            for (int j = 0; j < 8; ++j) { float d = y[j] - mu2; vs += d * d; }
            float rs2 = rsqrtf(waveSum(vs) * (1.0f / 256.0f) + 1e-5f);

            unsigned pk[4];
            #pragma unroll
            for (int jj = 0; jj < 4; ++jj) {
                float m0 = (y[2 * jj]     - mu2) * rs2 * gv2[2 * jj]     + bv2[2 * jj];
                float m1 = (y[2 * jj + 1] - mu2) * rs2 * gv2[2 * jj + 1] + bv2[2 * jj + 1];
                pk[jj] = (unsigned)f2bf(m0) | ((unsigned)f2bf(m1) << 16);
            }
            int mt    = mr >> 4;
            int laneP = (mr & 15) + 16 * hiL;
            uint4* dst = (uint4*)((char*)mBuf +
                         ((size_t)((kt * 8 + mt) * 512 + laneP * 16 + e0) * 2));
            uint4 v; v.x = pk[0]; v.y = pk[1]; v.z = pk[2]; v.w = pk[3];
            *dst = v;
        }
    }
    __threadfence();          // make y1 stores visible for Phase C readers
    __syncthreads();

    // ---------------- Phase B: GEMM1 (K=256) + GELU -> LDS h-fragments ------
    const v16bf* mV  = (const v16bf*)mBuf;
    const v16bf* w1V = (const v16bf*)w1p;
    for (int mt = 0; mt < 8; ++mt) {
        v16bf A[8];
        #pragma unroll
        for (int kt = 0; kt < 8; ++kt)
            A[kt] = mV[(kt * 8 + mt) * 32 + lane];

        for (int t = 0; t < 8; ++t) {
            int nt = wv * 8 + t;
            __builtin_prefetch(&w1V[((size_t)(nt + 1) & 63) * 32 + lane], 0, 1);
            float bias = b1[nt * 16 + (lane & 15)];
            v8f acc = {bias, bias, bias, bias, bias, bias, bias, bias};
            #pragma unroll
            for (int kt = 0; kt < 8; ++kt) {
                v16bf Bf = w1V[((size_t)(kt * 64 + nt)) * 32 + lane];
                acc = __builtin_amdgcn_wmma_f32_16x16x32_bf16(
                          false, A[kt], false, Bf, (short)0, acc, false, false);
            }
            // exact GELU (branch-free), scatter into A-fragment layout for GEMM2
            int K2  = nt * 16 + (lane & 15);
            int kt2 = K2 >> 5, kk2 = K2 & 31;
            int hi  = (kk2 >> 3) & 1;
            int e   = (kk2 & 7) + ((kk2 & 16) >> 1);
            int lp8 = (lane & 16) ? 8 : 0;
            #pragma unroll
            for (int r = 0; r < 8; ++r) {
                float gel = gelu_exact(acc[r]);
                int laneP = r + lp8 + 16 * hi;
                hBuf[(size_t)(kt2 * 8 + mt) * 512 + laneP * 16 + e] = f2bf(gel);
            }
        }
    }
    __syncthreads();

    // ---------------- Phase C: GEMM2 (K=1024) + bias + residual -------------
    const v16bf* hV  = (const v16bf*)hBuf;
    const v16bf* w2V = (const v16bf*)w2p;
    for (int mt = 0; mt < 8; ++mt) {
        for (int t = 0; t < 2; ++t) {
            int nt = wv * 2 + t;
            float bias = b2[nt * 16 + (lane & 15)];
            v8f acc = {bias, bias, bias, bias, bias, bias, bias, bias};
            #pragma unroll 8
            for (int kt = 0; kt < 32; ++kt) {
                v16bf Af = hV[(size_t)(kt * 8 + mt) * 32 + lane];
                v16bf Bf = w2V[((size_t)(kt * 16 + nt)) * 32 + lane];
                acc = __builtin_amdgcn_wmma_f32_16x16x32_bf16(
                          false, Af, false, Bf, (short)0, acc, false, false);
            }
            int col = nt * 16 + (lane & 15);
            int lp8 = (lane & 16) ? 8 : 0;
            #pragma unroll
            for (int r = 0; r < 8; ++r) {
                size_t idx = (size_t)(blk0 + mt * 16 + r + lp8) * C_DIM + col;
                out[idx] = out[idx] + acc[r];   // out held y1 -> final result
            }
        }
    }
}

// ---------------------------------------------------------------------------
extern "C" void kernel_launch(void* const* d_in, const int* in_sizes, int n_in,
                              void* d_out, int out_size, void* d_ws, size_t ws_size,
                              hipStream_t stream) {
    const float* x    = (const float*)d_in[0];
    const float* g1   = (const float*)d_in[1];
    const float* bt1  = (const float*)d_in[2];
    const float* g2   = (const float*)d_in[3];
    const float* bt2  = (const float*)d_in[4];
    const float* w1   = (const float*)d_in[5];
    const float* b1   = (const float*)d_in[6];
    const float* w2   = (const float*)d_in[7];
    const float* b2   = (const float*)d_in[8];
    float* out = (float*)d_out;

    unsigned short* w1p = (unsigned short*)d_ws;                           // 512KB
    unsigned short* w2p = (unsigned short*)((char*)d_ws + 256 * 1024 * 2); // 512KB

    // 512 tiles * 32 lanes = 16384 threads each
    pack_w_kernel<<<64, 256, 0, stream>>>(w1, w1p, C_DIM, HID);
    pack_w_kernel<<<64, 256, 0, stream>>>(w2, w2p, HID, C_DIM);

    swin_mlp_kernel<<<NROWS / MBLK, 256, LDS_BYTES, stream>>>(
        x, g1, bt1, g2, bt2, w1p, b1, w2p, b2, out);
}
